// SlotAttention_71640054497428
// MI455X (gfx1250) — matrix operand order repaired and checked by reading, not compiled, not run on hardware
//
#include <hip/hip_runtime.h>
#include <cstdint>
#include <cstddef>

// ============================================================================
// Slot-Attention forward for MI455X (gfx1250), bf16 WMMA everywhere.
//   C=512, HID=1024, B=64, S=8 slots, ITERS=3, N_img=1024, N_aud=512
// All GEMMs use v_wmma_f32_16x16x32_bf16 (K=32/issue -> 8x the f32 WMMA rate).
// attn@V stages V tiles through LDS (transposed) for the K-major B operand.
// q and attn are kept in zero-padded [*,16,K] bf16 layouts so the M=8 WMMA
// kernels are completely branchless in their hot loops.
// ============================================================================

typedef __bf16 bf16;
typedef __bf16 bf16x16 __attribute__((ext_vector_type(16)));
typedef float  f32x8   __attribute__((ext_vector_type(8)));
typedef float  f32x4   __attribute__((ext_vector_type(4)));

union Frag { bf16x16 v; f32x4 f4[2]; };

__device__ __forceinline__ f32x8 wmma_bf16(Frag a, Frag b, f32x8 c) {
  return __builtin_amdgcn_wmma_f32_16x16x32_bf16(
      /*neg_a=*/false, a.v, /*neg_b=*/false, b.v,
      /*c_mod=*/(short)0, c, /*reuse_a=*/false, /*reuse_b=*/false);
}

// A fragment (16x32 bf16, row-major source):
// lane l holds row m=l&15; one contiguous run of 16 K-values at kb=16*(l>=16).
__device__ __forceinline__ Frag load_a_frag(const bf16* rowPtr, int k) {
  Frag f;
  f.f4[0] = *(const f32x4*)(rowPtr + k);
  f.f4[1] = *(const f32x4*)(rowPtr + k + 8);
  return f;
}
// B fragment (32x16 bf16): lane l holds column n=l&15; one contiguous run of
// 16 K-values at kb=16*(l>=16)  (rowPtr pre-offset by n*ldk + 16*(l>=16)).
__device__ __forceinline__ Frag load_b_frag(const bf16* rowPtr, int k) {
  Frag f;
  f.f4[0] = *(const f32x4*)(rowPtr + k);
  f.f4[1] = *(const f32x4*)(rowPtr + k + 8);
  return f;
}

// ---------------------------------------------------------------------------
// Generic D[M,N] = act(A[M,K] * W[N,K]^T + bias[N]); A,W bf16; out f32/bf16.
// Block = 256 thr (8 waves) -> 128x64 tile; wave = 32x32 (4 WMMA tiles).
// M%128==0, N%64==0, K%64==0 for every call site in this pipeline.
// PAD: bf16 output row index remapped  m=b*8+s -> b*16+s  (rows 8..15 owned
// by a one-time zero fill) so downstream M=8 WMMA consumers are branchless.
// ---------------------------------------------------------------------------
template <int ACT, int PAD>
__global__ __launch_bounds__(256)
void gemm_bf16_kernel(const bf16* __restrict__ A, const bf16* __restrict__ W,
                      const float* __restrict__ bias,
                      float* __restrict__ outF, bf16* __restrict__ outB,
                      int M, int N, int K) {
  const int lane  = threadIdx.x & 31;
  const int wave  = threadIdx.x >> 5;
  const int wm    = wave & 3;            // 4 waves along M
  const int wn    = wave >> 2;           // 2 waves along N
  const int m0    = blockIdx.y * 128 + wm * 32;
  const int n0    = blockIdx.x * 64  + wn * 32;
  const int lm    = lane & 15;
  const int lh    = lane >> 4;

  const bf16* A0 = A + (size_t)(m0 + lm)      * K + 16 * lh;
  const bf16* A1 = A + (size_t)(m0 + 16 + lm) * K + 16 * lh;
  const bf16* B0 = W + (size_t)(n0 + lm)      * K + 16 * lh;
  const bf16* B1 = W + (size_t)(n0 + 16 + lm) * K + 16 * lh;

  f32x8 acc[2][2] = {};
#pragma unroll 2
  for (int k = 0; k < K; k += 32) {
    if (k + 64 < K) {                     // -> global_prefetch_b8
      __builtin_prefetch(A0 + k + 64, 0, 0);
      __builtin_prefetch(B0 + k + 64, 0, 0);
    }
    Frag a0 = load_a_frag(A0, k), a1 = load_a_frag(A1, k);
    Frag b0 = load_b_frag(B0, k), b1 = load_b_frag(B1, k);
    acc[0][0] = wmma_bf16(a0, b0, acc[0][0]);
    acc[0][1] = wmma_bf16(a0, b1, acc[0][1]);
    acc[1][0] = wmma_bf16(a1, b0, acc[1][0]);
    acc[1][1] = wmma_bf16(a1, b1, acc[1][1]);
  }

  // C/D layout: vgpr r, lanes 0-15 -> M=r, lanes 16-31 -> M=r+8; N=lane&15.
  for (int mi = 0; mi < 2; ++mi)
    for (int ni = 0; ni < 2; ++ni) {
      const int n  = n0 + ni * 16 + lm;
      const float bv = bias ? bias[n] : 0.0f;
#pragma unroll
      for (int r = 0; r < 8; ++r) {
        const int m = m0 + mi * 16 + (lh ? r + 8 : r);
        float val = acc[mi][ni][r] + bv;
        if (ACT) val = val > 0.0f ? val : 0.0f;
        if (outF) outF[(size_t)m * N + n] = val;
        if (outB) {
          const size_t mo = PAD ? ((size_t)(m >> 3) * 16 + (m & 7)) : (size_t)m;
          outB[mo * N + n] = (bf16)val;
        }
      }
    }
}

// ---------------------------------------------------------------------------
// dots[b, i<8, j] = scale * q[b,i,:] . Kmat[b,j,:]
// Q comes pre-padded [64,16,512] with zero rows 8..15 -> branchless WMMA loop.
// Block = 128 thr (4 waves); block tile = 16 x 64 keys; grid = (Nk/64, 64).
// ---------------------------------------------------------------------------
__global__ __launch_bounds__(128)
void dots_kernel(const bf16* __restrict__ Qpad, const bf16* __restrict__ Kmat,
                 float* __restrict__ dots, int Nk, float scale) {
  const int b    = blockIdx.y;
  const int lane = threadIdx.x & 31;
  const int wave = threadIdx.x >> 5;
  const int n0   = blockIdx.x * 64 + wave * 16;
  const int lm   = lane & 15;
  const int lh   = lane >> 4;

  const bf16* Qrow = Qpad + ((size_t)b * 16 + lm) * 512 + 16 * lh;
  const bf16* Krow = Kmat + ((size_t)b * Nk + n0 + lm) * 512 + 16 * lh;

  f32x8 acc = {};
#pragma unroll 2
  for (int k = 0; k < 512; k += 32) {
    Frag a  = load_a_frag(Qrow, k);
    Frag bb = load_b_frag(Krow, k);
    acc = wmma_bf16(a, bb, acc);
  }
  if (lh == 0) {                               // M = r in [0,8)
#pragma unroll
    for (int r = 0; r < 8; ++r)
      dots[((size_t)b * 8 + r) * Nk + n0 + lm] = acc[r] * scale;
  }
}

// ---------------------------------------------------------------------------
// attn = softmax(dots, slot axis) + eps, renormalized over keys.
// Writes f32 attn and a [16,Nk]-strided bf16 copy (pad rows pre-zeroed).
// ---------------------------------------------------------------------------
__global__ __launch_bounds__(256)
void inv_softmax_kernel(const float* __restrict__ dots, float* __restrict__ outF,
                        bf16* __restrict__ outPad, int Nk) {
  __shared__ float red[8][8];
  __shared__ float tot[8];
  const int b = blockIdx.x;
  const float* D = dots + (size_t)b * 8 * Nk;

  float part[8];
#pragma unroll
  for (int i = 0; i < 8; ++i) part[i] = 0.0f;

  for (int j = threadIdx.x; j < Nk; j += 256) {
    float e[8], mx = -1e30f;
#pragma unroll
    for (int i = 0; i < 8; ++i) { e[i] = D[(size_t)i * Nk + j]; mx = fmaxf(mx, e[i]); }
    float s = 0.0f;
#pragma unroll
    for (int i = 0; i < 8; ++i) { e[i] = __expf(e[i] - mx); s += e[i]; }
    const float inv = 1.0f / s;
#pragma unroll
    for (int i = 0; i < 8; ++i) part[i] += e[i] * inv + 1e-8f;
  }
  const int lane = threadIdx.x & 31, wave = threadIdx.x >> 5;
#pragma unroll
  for (int i = 0; i < 8; ++i) {
    float p = part[i];
    for (int o = 16; o; o >>= 1) p += __shfl_xor(p, o, 32);   // wave32
    if (lane == 0) red[i][wave] = p;
  }
  __syncthreads();
  if (threadIdx.x < 8) {
    float t = 0.0f;
    for (int w = 0; w < 8; ++w) t += red[threadIdx.x][w];
    tot[threadIdx.x] = 1.0f / t;
  }
  __syncthreads();

  float* O = outF + (size_t)b * 8 * Nk;
  for (int j = threadIdx.x; j < Nk; j += 256) {
    float e[8], mx = -1e30f;
#pragma unroll
    for (int i = 0; i < 8; ++i) { e[i] = D[(size_t)i * Nk + j]; mx = fmaxf(mx, e[i]); }
    float s = 0.0f;
#pragma unroll
    for (int i = 0; i < 8; ++i) { e[i] = __expf(e[i] - mx); s += e[i]; }
    const float inv = 1.0f / s;
#pragma unroll
    for (int i = 0; i < 8; ++i) {
      const float a = (e[i] * inv + 1e-8f) * tot[i];
      O[(size_t)i * Nk + j] = a;
      if (outPad) outPad[((size_t)b * 16 + i) * Nk + j] = (bf16)a;
    }
  }
}

// ---------------------------------------------------------------------------
// updates[b, i<8, d] = sum_j attn[b,i,j] * V[b,j,d]   (WMMA via LDS transpose)
// Block = 128 thr (4 waves) -> 16 x 64 output tile; grid = (512/64, 64).
// V tile [32k x 64d] staged transposed into LDS (row stride 96B, b128-aligned).
// ---------------------------------------------------------------------------
__global__ __launch_bounds__(128)
void updates_kernel(const bf16* __restrict__ attnPad, const bf16* __restrict__ V,
                    bf16* __restrict__ upd, int Nk) {
  __shared__ __align__(16) bf16 vt[64][48];
  const int b    = blockIdx.y;
  const int d0   = blockIdx.x * 64;
  const int lane = threadIdx.x & 31;
  const int wave = threadIdx.x >> 5;
  const int lm   = lane & 15;
  const int lh   = lane >> 4;

  const bf16* Arow = attnPad + ((size_t)b * 16 + lm) * Nk + 16 * lh;

  f32x8 acc = {};
  for (int k0 = 0; k0 < Nk; k0 += 32) {
    for (int idx = threadIdx.x; idx < 2048; idx += 128) {
      const int kk = idx >> 6, dd = idx & 63;
      vt[dd][kk] = V[((size_t)b * Nk + k0 + kk) * 512 + d0 + dd];
    }
    __syncthreads();
    Frag a = load_a_frag(Arow, k0);
    Frag bb;
    const bf16* brow = &vt[wave * 16 + lm][16 * lh];
    bb.f4[0] = *(const f32x4*)brow;            // ds_load_b128
    bb.f4[1] = *(const f32x4*)(brow + 8);
    acc = wmma_bf16(a, bb, acc);
    __syncthreads();
  }
  if (lh == 0) {
    const int d = d0 + wave * 16 + lm;
#pragma unroll
    for (int r = 0; r < 8; ++r)
      upd[((size_t)b * 8 + r) * 512 + d] = (bf16)acc[r];
  }
}

// ---------------------------------------------------------------------------
// Row LayerNorm over 512 (one wave per row), bf16 out. rows % 8 == 0.
// ---------------------------------------------------------------------------
__global__ __launch_bounds__(256)
void ln_kernel(const float* __restrict__ x, const float* __restrict__ g,
               const float* __restrict__ beta, bf16* __restrict__ out, int rows) {
  const int wave = threadIdx.x >> 5, lane = threadIdx.x & 31;
  const int row  = blockIdx.x * 8 + wave;
  if (row >= rows) return;
  const float* xr = x + (size_t)row * 512;
  f32x4 v[4];
  float s = 0.0f, s2 = 0.0f;
#pragma unroll
  for (int i = 0; i < 4; ++i) {
    v[i] = *(const f32x4*)(xr + lane * 16 + i * 4);
    s  += v[i].x + v[i].y + v[i].z + v[i].w;
    s2 += v[i].x * v[i].x + v[i].y * v[i].y + v[i].z * v[i].z + v[i].w * v[i].w;
  }
  for (int o = 16; o; o >>= 1) { s += __shfl_xor(s, o, 32); s2 += __shfl_xor(s2, o, 32); }
  const float mean = s * (1.0f / 512.0f);
  const float var  = s2 * (1.0f / 512.0f) - mean * mean;
  const float rs   = rsqrtf(var + 1e-5f);
  bf16* orow = out + (size_t)row * 512;
#pragma unroll
  for (int i = 0; i < 4; ++i) {
    const int c = lane * 16 + i * 4;
    orow[c + 0] = (bf16)((v[i].x - mean) * rs * g[c + 0] + beta[c + 0]);
    orow[c + 1] = (bf16)((v[i].y - mean) * rs * g[c + 1] + beta[c + 1]);
    orow[c + 2] = (bf16)((v[i].z - mean) * rs * g[c + 2] + beta[c + 2]);
    orow[c + 3] = (bf16)((v[i].w - mean) * rs * g[c + 3] + beta[c + 3]);
  }
}

// ------------------------- small elementwise kernels -----------------------
__global__ void cvt_bf16_kernel(const float* __restrict__ in, bf16* __restrict__ out, int n) {
  const int i = blockIdx.x * 256 + threadIdx.x;
  if (i < n) out[i] = (bf16)in[i];
}

__global__ void zero_bf16_kernel(bf16* __restrict__ p, int n) {
  const int i = blockIdx.x * 256 + threadIdx.x;
  if (i < n) p[i] = (bf16)0.0f;
}

__global__ void bcast_slots_kernel(const float* __restrict__ s0,
                                   float* __restrict__ outF, bf16* __restrict__ outB) {
  const int i = blockIdx.x * 256 + threadIdx.x;      // 0..262143
  const float v = s0[i & 4095];
  outF[i] = v; outB[i] = (bf16)v;
}

__global__ void gru_kernel(const float* __restrict__ gi, const float* __restrict__ gh,
                           const float* __restrict__ prev, float* __restrict__ h) {
  const int i   = blockIdx.x * 256 + threadIdx.x;    // 0..262143
  const int row = i >> 9, c = i & 511;
  const float* giR = gi + (size_t)row * 1536;
  const float* ghR = gh + (size_t)row * 1536;
  const float r = 1.0f / (1.0f + __expf(-(giR[c] + ghR[c])));
  const float z = 1.0f / (1.0f + __expf(-(giR[512 + c] + ghR[512 + c])));
  const float n = tanhf(giR[1024 + c] + r * ghR[1024 + c]);
  h[i] = (1.0f - z) * n + z * prev[i];
}

__global__ void add_ff_kernel(const float* __restrict__ h, const float* __restrict__ ff,
                              float* __restrict__ outF, bf16* __restrict__ outB) {
  const int i = blockIdx.x * 256 + threadIdx.x;
  const float v = h[i] + ff[i];
  outF[i] = v; outB[i] = (bf16)v;
}

// ============================================================================
// Host orchestration
// ============================================================================
extern "C" void kernel_launch(void* const* d_in, const int* in_sizes, int n_in,
                              void* d_out, int out_size, void* d_ws, size_t ws_size,
                              hipStream_t stream) {
  (void)in_sizes; (void)n_in; (void)out_size; (void)ws_size;

  const float* img    = (const float*)d_in[0];   // [64,1024,512]
  const float* aud    = (const float*)d_in[1];   // [64,512,512]
  const float* slots0 = (const float*)d_in[2];   // [1,8,512]
  // params per modality, insertion order:
  // 0 qW 1 qb 2 kW 3 kb 4 vW 5 vb 6 Wih 7 Whh 8 bih 9 bhh
  // 10 W1 11 b1 12 W2 13 b2 14 ln_in_g 15 ln_in_b 16 ln_sl_g 17 ln_sl_b 18 ln_ff_g 19 ln_ff_b
  auto P = [&](int m, int i) -> const float* { return (const float*)d_in[3 + m * 20 + i]; };

  // ---- workspace carve-out ----
  char* base = (char*)d_ws;
  size_t cur = 0;
  auto take = [&](size_t bytes) -> void* {
    cur = (cur + 255) & ~(size_t)255;
    void* p = base + cur; cur += bytes; return p;
  };
  bf16* imgX = (bf16*)take((size_t)64 * 1024 * 512 * 2);
  bf16* audX = (bf16*)take((size_t)64 * 512  * 512 * 2);
  bf16* imgK = (bf16*)take((size_t)64 * 1024 * 512 * 2);
  bf16* imgV = (bf16*)take((size_t)64 * 1024 * 512 * 2);
  bf16* audK = (bf16*)take((size_t)64 * 512  * 512 * 2);
  bf16* audV = (bf16*)take((size_t)64 * 512  * 512 * 2);
  struct WB { bf16 *qW, *kW, *vW, *Wih, *Whh, *W1, *W2; } wb[2];
  for (int m = 0; m < 2; ++m) {
    wb[m].qW  = (bf16*)take(262144 * 2);  wb[m].kW  = (bf16*)take(262144 * 2);
    wb[m].vW  = (bf16*)take(262144 * 2);  wb[m].Wih = (bf16*)take(786432 * 2);
    wb[m].Whh = (bf16*)take(786432 * 2);  wb[m].W1  = (bf16*)take(524288 * 2);
    wb[m].W2  = (bf16*)take(524288 * 2);
  }
  bf16* slotsB[2] = { (bf16*)take(262144 * 2), (bf16*)take(262144 * 2) };
  bf16*  snB     = (bf16*)take(262144 * 2);
  bf16*  qPad    = (bf16*)take((size_t)64 * 16 * 512 * 2);   // zero rows 8..15
  float* dotsBuf = (float*)take((size_t)64 * 8 * 1024 * 4);
  bf16*  attnPad = (bf16*)take((size_t)64 * 16 * 1024 * 2);  // zero rows 8..15
  bf16*  updB    = (bf16*)take(262144 * 2);
  float* gi      = (float*)take(786432 * 4);
  float* gh      = (float*)take(786432 * 4);
  float* hBuf    = (float*)take(262144 * 4);
  bf16*  hnB     = (bf16*)take(262144 * 2);
  bf16*  ff1B    = (bf16*)take(524288 * 2);
  float* ff2     = (float*)take(262144 * 4);

  // ---- output regions (all f32, concatenated in tuple order) ----
  float* out = (float*)d_out;
  float* outImgSlots = out;                // [64,8,512]
  float* outAudSlots = out + 262144;
  float* outImgQ     = out + 524288;
  float* outAudQ     = out + 786432;
  float* outAII      = out + 1048576;      // [64,8,1024]
  float* outAIA      = out + 1572864;      // [64,8,512]
  float* outAAA      = out + 1835008;      // [64,8,512]
  float* outAAI      = out + 2097152;      // [64,8,1024]

  const float SCALE_F = 0.04419417382415922f;   // 512^-0.5

  // ---- phase 0: weight casts, input LN, slot broadcast, pad zero-fill ----
  for (int m = 0; m < 2; ++m) {
    cvt_bf16_kernel<<<1024, 256, 0, stream>>>(P(m, 0),  wb[m].qW,  262144);
    cvt_bf16_kernel<<<1024, 256, 0, stream>>>(P(m, 2),  wb[m].kW,  262144);
    cvt_bf16_kernel<<<1024, 256, 0, stream>>>(P(m, 4),  wb[m].vW,  262144);
    cvt_bf16_kernel<<<3072, 256, 0, stream>>>(P(m, 6),  wb[m].Wih, 786432);
    cvt_bf16_kernel<<<3072, 256, 0, stream>>>(P(m, 7),  wb[m].Whh, 786432);
    cvt_bf16_kernel<<<2048, 256, 0, stream>>>(P(m, 10), wb[m].W1,  524288);
    cvt_bf16_kernel<<<2048, 256, 0, stream>>>(P(m, 12), wb[m].W2,  524288);
  }
  zero_bf16_kernel<<<2048, 256, 0, stream>>>(qPad, 524288);  // once: rows 8..15 stay 0
  ln_kernel<<<8192, 256, 0, stream>>>(img, P(0, 14), P(0, 15), imgX, 65536);
  ln_kernel<<<4096, 256, 0, stream>>>(aud, P(1, 14), P(1, 15), audX, 32768);
  bcast_slots_kernel<<<1024, 256, 0, stream>>>(slots0, outImgSlots, slotsB[0]);
  bcast_slots_kernel<<<1024, 256, 0, stream>>>(slots0, outAudSlots, slotsB[1]);

  // ---- phase 1: K/V projections (the big WMMA GEMMs) ----
  gemm_bf16_kernel<0,0><<<dim3(8, 512), 256, 0, stream>>>(imgX, wb[0].kW, P(0, 3), nullptr, imgK, 65536, 512, 512);
  gemm_bf16_kernel<0,0><<<dim3(8, 512), 256, 0, stream>>>(imgX, wb[0].vW, P(0, 5), nullptr, imgV, 65536, 512, 512);
  gemm_bf16_kernel<0,0><<<dim3(8, 256), 256, 0, stream>>>(audX, wb[1].kW, P(1, 3), nullptr, audK, 32768, 512, 512);
  gemm_bf16_kernel<0,0><<<dim3(8, 256), 256, 0, stream>>>(audX, wb[1].vW, P(1, 5), nullptr, audV, 32768, 512, 512);

  // ---- phase 2/3: slot iterations per modality ----
  auto run_slots = [&](int m, const bf16* selfK, int Nself, const bf16* crossK, int Ncross,
                       const bf16* selfV, float* outSlots, float* outQ,
                       float* outAttn, float* outCattn) {
    // zero whole attnPad once per modality (Nk stride differs across them);
    // inv_softmax then only ever writes rows 0..7 of the [16,Nk] layout.
    zero_bf16_kernel<<<4096, 256, 0, stream>>>(attnPad, 1048576);
    for (int it = 0; it < 3; ++it) {
      ln_kernel<<<64, 256, 0, stream>>>(outSlots, P(m, 16), P(m, 17), snB, 512);
      gemm_bf16_kernel<0,1><<<dim3(8, 4), 256, 0, stream>>>(snB, wb[m].qW, P(m, 1), outQ, qPad, 512, 512, 512);
      dots_kernel<<<dim3(Nself / 64, 64), 128, 0, stream>>>(qPad, selfK, dotsBuf, Nself, SCALE_F);
      inv_softmax_kernel<<<64, 256, 0, stream>>>(dotsBuf, outAttn, attnPad, Nself);
      dots_kernel<<<dim3(Ncross / 64, 64), 128, 0, stream>>>(qPad, crossK, dotsBuf, Ncross, SCALE_F);
      inv_softmax_kernel<<<64, 256, 0, stream>>>(dotsBuf, outCattn, nullptr, Ncross);
      updates_kernel<<<dim3(8, 64), 128, 0, stream>>>(attnPad, selfV, updB, Nself);
      gemm_bf16_kernel<0,0><<<dim3(24, 4), 256, 0, stream>>>(updB,      wb[m].Wih, P(m, 8), gi, nullptr, 512, 1536, 512);
      gemm_bf16_kernel<0,0><<<dim3(24, 4), 256, 0, stream>>>(slotsB[m], wb[m].Whh, P(m, 9), gh, nullptr, 512, 1536, 512);
      gru_kernel<<<1024, 256, 0, stream>>>(gi, gh, outSlots, hBuf);
      ln_kernel<<<64, 256, 0, stream>>>(hBuf, P(m, 18), P(m, 19), hnB, 512);
      gemm_bf16_kernel<1,0><<<dim3(16, 4), 256, 0, stream>>>(hnB,  wb[m].W1, P(m, 11), nullptr, ff1B, 512, 1024, 512);
      gemm_bf16_kernel<0,0><<<dim3(8, 4),  256, 0, stream>>>(ff1B, wb[m].W2, P(m, 13), ff2, nullptr, 512, 512, 1024);
      add_ff_kernel<<<1024, 256, 0, stream>>>(hBuf, ff2, outSlots, slotsB[m]);
    }
  };
  run_slots(0, imgK, 1024, audK, 512,  imgV, outImgSlots, outImgQ, outAII, outAIA);
  run_slots(1, audK, 512,  imgK, 1024, audV, outAudSlots, outAudQ, outAAA, outAAI);
}
